// InfoNCELoss_58085137711484
// MI455X (gfx1250) — matrix operand order repaired and checked by reading, not compiled, not run on hardware
//
#include <hip/hip_runtime.h>
#include <hip/hip_fp16.h>

typedef __attribute__((ext_vector_type(16))) _Float16 v16h;
typedef __attribute__((ext_vector_type(8)))  _Float16 v8h;
typedef __attribute__((ext_vector_type(8)))  float    v8f;

#define N_EMB 8192
#define D     512
#define N_POS 4096
#define N_NEG 16384
#define INV_T (1.0f / 0.07f)
// Fixed logsumexp shift: cosine sims <= 1, so max scaled logit <= INV_T.
// lse = INV_T + log( sum exp((sim - 1) * INV_T) )  -- exact shift identity,
// argument bounded in [-2/T, 0] -> no overflow, benign underflow.

#define M_TILE   64          // anchors per block (A tile in LDS)
#define N_SPLIT  8           // N-dimension split across blockIdx.y

// ---------------------------------------------------------------------------
// Kernel 0: zero the per-row exp-sum accumulators + metric accumulators.
// ---------------------------------------------------------------------------
__global__ void init_kernel(float* __restrict__ g_s, float* __restrict__ accum) {
  const int gid = blockIdx.x * blockDim.x + threadIdx.x;
  if (gid < N_POS) g_s[gid] = 0.f;
  if (gid < 2) accum[gid] = 0.f;
}

// ---------------------------------------------------------------------------
// Kernel 1: L2-normalize embeddings, store as f16 (one wave per row).
// ---------------------------------------------------------------------------
__global__ __launch_bounds__(256)
void normalize_kernel(const float* __restrict__ emb, _Float16* __restrict__ e16) {
  const int wave = (blockIdx.x * blockDim.x + threadIdx.x) >> 5;
  const int lane = threadIdx.x & 31;
  if (wave >= N_EMB) return;

  const float* row = emb + (size_t)wave * D + lane * 16;
  float v[16];
  float ss = 0.f;
#pragma unroll
  for (int i = 0; i < 16; ++i) { v[i] = row[i]; ss += v[i] * v[i]; }
#pragma unroll
  for (int off = 16; off; off >>= 1) ss += __shfl_xor(ss, off, 32);

  const float inv = 1.0f / fmaxf(sqrtf(ss), 1e-8f);
  v16h o;
#pragma unroll
  for (int i = 0; i < 16; ++i) o[i] = (_Float16)(v[i] * inv);
  *(v16h*)(e16 + (size_t)wave * D + lane * 16) = o;
}

// ---------------------------------------------------------------------------
// Kernel 2: gathered GEMM, M_TILE=64 anchors/block x 1024-column N-slice,
// K=512, via v_wmma_f32_16x16x32_f16.  Each B fragment feeds 4 WMMAs
// (4 M-subtiles); A stays in LDS (ds_load_b128) -- an opaque per-iteration
// offset pins the A loads inside the nt loop so LICM cannot promote the
// 512-VGPR A working set into (spilled) registers.
// Epilogue: sum of exp((sim-1)/T) per row, atomically added per block.
// ---------------------------------------------------------------------------
__global__ __launch_bounds__(256)
void infonce_rows_kernel(const _Float16* __restrict__ e16,
                         const int* __restrict__ pos_anchor,
                         float* __restrict__ g_s) {
  __shared__ __align__(32) _Float16 As[M_TILE * D];   // 64 KB
  __shared__ float red_s[8][M_TILE];                  // 2 KB

  const int mblk = blockIdx.x;        // 64 M-blocks of 64 anchors
  const int nblk = blockIdx.y;        // 8 N-slices of 64 N-tiles each
  const int tid  = threadIdx.x;
  const int wave = tid >> 5;          // 8 waves
  const int lane = tid & 31;

  // Stage 64 gathered anchor rows into LDS: 256 B per thread.
  {
    const int row  = tid >> 2;              // 0..63
    const int colb = (tid & 3) * 128;       // 0..384
    const int a    = pos_anchor[mblk * M_TILE + row];
    const v16h* src = (const v16h*)(e16 + (size_t)a * D + colb);
    v16h*       dst = (v16h*)(As + row * D + colb);
#pragma unroll
    for (int i = 0; i < 8; ++i) dst[i] = src[i];
  }
  __syncthreads();

  const int khalf = lane >> 4;   // lane half: selects K sub-chunk (A) / K-half (B)
  const int mrow  = lane & 15;   // A row within subtile == B column within tile

  float sacc[4][8];
#pragma unroll
  for (int t = 0; t < 4; ++t)
#pragma unroll
    for (int i = 0; i < 8; ++i) sacc[t][i] = 0.f;

  for (int nt = wave; nt < 64; nt += 8) {
    // Opaque zero: A-load addresses depend on a per-iteration asm result,
    // so the ds_load_b128s cannot be hoisted/promoted out of this loop.
    int zoff;
    asm volatile("s_mov_b32 %0, 0" : "=s"(zoff));
    const _Float16* Abase = As + zoff + mrow * D + khalf * 16;

    const int col0 = (nblk * 64 + nt) * 16;
    const _Float16* bcol = e16 + (size_t)(col0 + mrow) * D + khalf * 16;
    if (nt + 8 < 64)
      __builtin_prefetch(bcol + (size_t)8 * 16 * D, 0, 3);

    v8f acc[4] = {};
#pragma unroll 4
    for (int kt = 0; kt < D / 32; ++kt) {
      const int kb = kt * 32;
      // B 32x16 f16: lane half h holds K = kb + 16h .. +15 of column mrow.
      const v16h b = *(const v16h*)(bcol + kb);
#pragma unroll
      for (int t = 0; t < 4; ++t) {
        // A 16x32 f16: lane half h holds K = kb + 16h .. +15.
        const _Float16* Ar = Abase + t * 16 * D + kb;
        v8h alo = *(const v8h*)(Ar);
        v8h ahi = *(const v8h*)(Ar + 8);
        v16h a  = __builtin_shufflevector(alo, ahi,
                    0, 1, 2, 3, 4, 5, 6, 7, 8, 9, 10, 11, 12, 13, 14, 15);
        acc[t] = __builtin_amdgcn_wmma_f32_16x16x32_f16(
                     false, a, false, b, (short)0, acc[t], false, false);
      }
    }

    // Epilogue: one exp + one add per value (fixed shift, no running max).
#pragma unroll
    for (int t = 0; t < 4; ++t)
#pragma unroll
      for (int i = 0; i < 8; ++i)
        sacc[t][i] += __expf((acc[t][i] - 1.0f) * INV_T);
  }

  // Reduce over the 16 lanes of each half (N dimension) -- pure adds.
#pragma unroll
  for (int t = 0; t < 4; ++t)
#pragma unroll
    for (int i = 0; i < 8; ++i) {
      float s = sacc[t][i];
#pragma unroll
      for (int off = 1; off < 16; off <<= 1) s += __shfl_xor(s, off, 32);
      if (mrow == 0)                        // lanes 0 and 16
        red_s[wave][t * 16 + khalf * 8 + i] = s;
    }
  __syncthreads();

  // Combine the 8 waves; one atomicAdd per row per block.
  if (tid < M_TILE) {
    float s = 0.f;
#pragma unroll
    for (int w = 0; w < 8; ++w) s += red_s[w][tid];
    atomicAdd(&g_s[mblk * M_TILE + tid], s);
  }
}

// ---------------------------------------------------------------------------
// Kernel 3: pairwise cosine dots (one wave per pair), mean via atomics.
// ---------------------------------------------------------------------------
__global__ __launch_bounds__(256)
void pair_dot_kernel(const _Float16* __restrict__ e16,
                     const int* __restrict__ ia, const int* __restrict__ ib,
                     int n, float* __restrict__ dots, float* __restrict__ accum) {
  const int gid  = blockIdx.x * blockDim.x + threadIdx.x;
  const int w    = gid >> 5;
  const int lane = gid & 31;
  if (w >= n) return;

  const v16h a = *(const v16h*)(e16 + (size_t)ia[w] * D + lane * 16);
  const v16h b = *(const v16h*)(e16 + (size_t)ib[w] * D + lane * 16);
  float s = 0.f;
#pragma unroll
  for (int i = 0; i < 16; ++i) s += (float)a[i] * (float)b[i];
#pragma unroll
  for (int off = 16; off; off >>= 1) s += __shfl_xor(s, off, 32);

  if (lane == 0) {
    atomicAdd(accum, s);
    if (dots) dots[w] = s;
  }
}

// ---------------------------------------------------------------------------
// Kernel 4: scalar finalization.  out = {loss, pos_sim, neg_sim}
// lse[p] = INV_T + log(g_s[p]);  pos_logit = posdot*INV_T.
// ---------------------------------------------------------------------------
__global__ __launch_bounds__(256)
void finalize_kernel(const float* __restrict__ g_s,
                     const float* __restrict__ posdot,
                     const float* __restrict__ accum,
                     float* __restrict__ out) {
  __shared__ float red[256];
  const int tid = threadIdx.x;
  float s = 0.f;
  for (int p = tid; p < N_POS; p += 256)
    s += (INV_T + __logf(g_s[p])) - posdot[p] * INV_T;
  red[tid] = s;
  __syncthreads();
  for (int off = 128; off; off >>= 1) {
    if (tid < off) red[tid] += red[tid + off];
    __syncthreads();
  }
  if (tid == 0) {
    out[0] = red[0] / (float)N_POS;
    out[1] = accum[0] / (float)N_POS;
    out[2] = accum[1] / (float)N_NEG;
  }
}

// ---------------------------------------------------------------------------
extern "C" void kernel_launch(void* const* d_in, const int* in_sizes, int n_in,
                              void* d_out, int out_size, void* d_ws, size_t ws_size,
                              hipStream_t stream) {
  (void)in_sizes; (void)n_in; (void)out_size; (void)ws_size;

  const float* emb = (const float*)d_in[0];
  const int*   pa  = (const int*)d_in[1];
  const int*   pt  = (const int*)d_in[2];
  const int*   na  = (const int*)d_in[3];
  const int*   ng  = (const int*)d_in[4];
  float*       out = (float*)d_out;

  char* w = (char*)d_ws;
  _Float16* e16    = (_Float16*)w;                              // 8 MB
  float*    g_s    = (float*)(w + (size_t)N_EMB * D * 2);       // 16 KB
  float*    posdot = g_s + N_POS;                               // 16 KB
  float*    accum  = posdot + N_POS;                            // 8 B

  init_kernel<<<N_POS / 256, 256, 0, stream>>>(g_s, accum);
  normalize_kernel<<<N_EMB / 8, 256, 0, stream>>>(emb, e16);
  infonce_rows_kernel<<<dim3(N_POS / M_TILE, N_SPLIT), 256, 0, stream>>>(e16, pa, g_s);
  pair_dot_kernel<<<(N_POS * 32 + 255) / 256, 256, 0, stream>>>(e16, pa, pt, N_POS, posdot, accum + 0);
  pair_dot_kernel<<<(N_NEG * 32 + 255) / 256, 256, 0, stream>>>(e16, na, ng, N_NEG, nullptr, accum + 1);
  finalize_kernel<<<1, 256, 0, stream>>>(g_s, posdot, accum, out);
}
